// GATModel_12515534701298
// MI455X (gfx1250) — matrix operand order repaired and checked by reading, not compiled, not run on hardware
//
#include <hip/hip_runtime.h>

typedef __attribute__((ext_vector_type(16))) _Float16 v16h;
typedef __attribute__((ext_vector_type(8)))  _Float16 v8h;
typedef __attribute__((ext_vector_type(8)))  float    v8f;

#define GN 50000
#define GE 800000
#define NCHUNK 64

// ---------------------------------------------------------------- utilities
__global__ void k_zero_i32(int* __restrict__ p, int n) {
  int i = blockIdx.x * blockDim.x + threadIdx.x;
  if (i < n) p[i] = 0;
}

__global__ void k_cvt_f16(const float* __restrict__ in, _Float16* __restrict__ out, int n) {
  int i = blockIdx.x * blockDim.x + threadIdx.x;
  if (i < n) out[i] = (_Float16)in[i];
}

// -------------------------------------------- deterministic CSR build by dst
// chunked stable counting sort: 64 chunks processed serially (one thread each)
__global__ void k_chunk_count(const int* __restrict__ dst, int* __restrict__ hist,
                              int* __restrict__ rank, int nEdges, int chunkSz, int nNodes) {
  int c = blockIdx.x * blockDim.x + threadIdx.x;
  if (c >= NCHUNK) return;
  int beg = c * chunkSz;
  int end = beg + chunkSz; if (end > nEdges) end = nEdges;
  int* h = hist + (long)c * nNodes;
  for (int e = beg; e < end; ++e) {
    int d = dst[e];
    int r = h[d];
    rank[e] = r;
    h[d] = r + 1;
  }
}

__global__ void k_node_deg(const int* __restrict__ hist, int* __restrict__ deg, int nNodes) {
  int d = blockIdx.x * blockDim.x + threadIdx.x;
  if (d >= nNodes) return;
  int s = 0;
  for (int c = 0; c < NCHUNK; ++c) s += hist[(long)c * nNodes + d];
  deg[d] = s;
}

__global__ void k_scan_excl(const int* __restrict__ deg, int* __restrict__ row_ptr, int n) {
  __shared__ int buf[1024];
  __shared__ int carry;
  if (threadIdx.x == 0) carry = 0;
  __syncthreads();
  for (int base = 0; base < n; base += 1024) {
    int i = base + threadIdx.x;
    int v = (i < n) ? deg[i] : 0;
    buf[threadIdx.x] = v;
    __syncthreads();
    for (int off = 1; off < 1024; off <<= 1) {
      int t = (threadIdx.x >= off) ? buf[threadIdx.x - off] : 0;
      __syncthreads();
      buf[threadIdx.x] += t;
      __syncthreads();
    }
    int cbase = carry;
    if (i < n) row_ptr[i] = cbase + buf[threadIdx.x] - v;
    __syncthreads();
    if (threadIdx.x == 0) carry = cbase + buf[1023];
    __syncthreads();
  }
  if (threadIdx.x == 0) row_ptr[n] = carry;
}

__global__ void k_chunk_offsets(int* __restrict__ hist, const int* __restrict__ row_ptr, int nNodes) {
  int d = blockIdx.x * blockDim.x + threadIdx.x;
  if (d >= nNodes) return;
  int running = row_ptr[d];
  for (int c = 0; c < NCHUNK; ++c) {
    long idx = (long)c * nNodes + d;
    int h = hist[idx];
    hist[idx] = running;
    running += h;
  }
}

__global__ void k_scatter_edges(const int* __restrict__ dst, const int* __restrict__ hist,
                                const int* __restrict__ rank, int* __restrict__ perm,
                                int nEdges, int chunkSz, int nNodes) {
  int e = blockIdx.x * blockDim.x + threadIdx.x;
  if (e >= nEdges) return;
  int c = e / chunkSz;
  perm[hist[(long)c * nNodes + dst[e]] + rank[e]] = e;
}

// ------------------------------------------------- B-operand fragment packer
// Packs W[K,Nc] (f32) into WMMA B-fragment order, fused with the f16 convert:
// Bp[((kt*tilesN + tn)*32 + lane)*16 + e] = (f16) W[(kt*32 + 16*(lane>>4) + e)*Nc
//                                                   + tn*16 + (lane&15)]
// so the GEMM reads each lane's fragment as one contiguous 32-byte run.
template<int K>
__global__ void k_pack_B(const float* __restrict__ W, _Float16* __restrict__ Bp, int Nc) {
  int idx = blockIdx.x * blockDim.x + threadIdx.x;   // (kt, tn, lane)
  int tilesN = Nc >> 4;
  int total = (K / 32) * tilesN * 32;
  if (idx >= total) return;
  int lane = idx & 31;
  int t2 = idx >> 5;
  int tn = t2 % tilesN;
  int kt = t2 / tilesN;
  int col = tn * 16 + (lane & 15);
  int k0  = kt * 32 + (lane >> 4) * 16;
  _Float16* o = Bp + (size_t)idx * 16;
#pragma unroll
  for (int e = 0; e < 16; ++e)
    o[e] = (_Float16)W[(size_t)(k0 + e) * Nc + col];
}

// -------------------------------------------------------------- WMMA GEMM
// C[M,Nc] = A[M,K] * Bpacked, f16 in / f32 out.
// One wave computes NTILE adjacent 16x16 tiles (A fragment reused NTILE times),
// with an explicit 1-deep software pipeline: operands of k-step kt+1 are
// prefetched before the WMMAs of k-step kt are issued.
__device__ __forceinline__ v16h load_a_frag(const _Float16* __restrict__ arow,
                                            int kt, int half) {
  return *(const v16h*)(arow + kt * 32 + half * 16);
}

template<int K, int NTILE>
__global__ void k_wmma_gemm_f16(const _Float16* __restrict__ A,
                                const _Float16* __restrict__ Bp,
                                float* __restrict__ C,
                                int M, int Nc) {
  constexpr int KT = K / 32;
  int wave = (blockIdx.x * blockDim.x + threadIdx.x) >> 5;
  int lane = threadIdx.x & 31;
  int tilesN  = Nc >> 4;
  int groupsN = tilesN / NTILE;
  int groups  = (M >> 4) * groupsN;
  if (wave >= groups) return;            // wave-uniform: EXEC stays all-ones
  int tm  = wave / groupsN;
  int tn0 = (wave % groupsN) * NTILE;
  int half = lane >> 4;
  int l15  = lane & 15;
  const _Float16* arow = A + (size_t)(tm * 16 + l15) * K;
  const _Float16* bbase = Bp + ((size_t)tn0 * 32 + lane) * 16;
  // lane-invariant stride between kt planes and between adjacent tn tiles
  const size_t bKtStride = (size_t)tilesN * 32 * 16;
  const size_t bTnStride = (size_t)32 * 16;

  v8f zero = {};
  v8f c[NTILE];
#pragma unroll
  for (int u = 0; u < NTILE; ++u) c[u] = zero;

  // prologue: fetch k-step 0 operands
  v16h a = load_a_frag(arow, 0, half);
  v16h b[NTILE];
#pragma unroll
  for (int u = 0; u < NTILE; ++u)
    b[u] = *(const v16h*)(bbase + u * bTnStride);

#pragma unroll
  for (int kt = 0; kt < KT; ++kt) {
    v16h an = a;
    v16h bn[NTILE];
    if (kt + 1 < KT) {                   // compile-time after full unroll
      an = load_a_frag(arow, kt + 1, half);
#pragma unroll
      for (int u = 0; u < NTILE; ++u)
        bn[u] = *(const v16h*)(bbase + (kt + 1) * bKtStride + u * bTnStride);
    }
#pragma unroll
    for (int u = 0; u < NTILE; ++u)
      c[u] = __builtin_amdgcn_wmma_f32_16x16x32_f16(false, a, false, b[u],
                                                    (short)0, c[u], false, false);
    if (kt + 1 < KT) {
      a = an;
#pragma unroll
      for (int u = 0; u < NTILE; ++u) b[u] = bn[u];
    }
  }

#pragma unroll
  for (int u = 0; u < NTILE; ++u)
#pragma unroll
    for (int j = 0; j < 8; ++j)
      C[(size_t)(tm * 16 + half * 8 + j) * Nc + (tn0 + u) * 16 + l15] = c[u][j];
}

// --------------------------------------------------------- attention scores
template<int H, int D>
__global__ void k_attn_scores(const float* __restrict__ feat,
                              const float* __restrict__ attn_l,
                              const float* __restrict__ attn_r,
                              float* __restrict__ el, float* __restrict__ er, int nNodes) {
  int idx = blockIdx.x * blockDim.x + threadIdx.x;
  if (idx >= nNodes * H) return;
  int node = idx / H, hh = idx % H;
  const float* f = feat + (size_t)node * H * D + (size_t)hh * D;
  float sl = 0.f, sr = 0.f;
#pragma unroll 4
  for (int d = 0; d < D; ++d) {
    float v = f[d];
    sl += v * attn_l[hh * D + d];
    sr += v * attn_r[hh * D + d];
  }
  el[idx] = sl;
  er[idx] = sr;
}

// ----------------------------------------- per-node softmax + aggregation
__device__ inline void store_val(float* p, float v) { *p = v; }
__device__ inline void store_val(_Float16* p, float v) { *p = (_Float16)v; }

template<int H, int D, bool ELU, typename OutT>
__global__ void k_aggregate(const float* __restrict__ feat,
                            const float* __restrict__ el, const float* __restrict__ er,
                            const int* __restrict__ row_ptr, const int* __restrict__ perm,
                            const int* __restrict__ srcv,
                            OutT* __restrict__ out, int nNodes) {
  constexpr int TOT = H * D;
  constexpr int PER = TOT / 32;           // 8 (layer1) or 4 (layer2) dims per lane
  int wave = (blockIdx.x * blockDim.x + threadIdx.x) >> 5;
  int lane = threadIdx.x & 31;
  if (wave >= nNodes) return;
  int beg = row_ptr[wave], end = row_ptr[wave + 1];

  float ern[H];
#pragma unroll
  for (int h = 0; h < H; ++h) ern[h] = er[wave * H + h];

  // pass 1: per-head max of leaky-relu scores (lane-parallel over edges)
  float mx[H];
#pragma unroll
  for (int h = 0; h < H; ++h) mx[h] = -3.402823466e38f;
  for (int i = beg + lane; i < end; i += 32) {
    int s = srcv[perm[i]];
#pragma unroll
    for (int h = 0; h < H; ++h) {
      float sc = el[s * H + h] + ern[h];
      sc = sc > 0.f ? sc : 0.2f * sc;
      mx[h] = fmaxf(mx[h], sc);
    }
  }
#pragma unroll
  for (int h = 0; h < H; ++h)
    for (int off = 16; off > 0; off >>= 1)
      mx[h] = fmaxf(mx[h], __shfl_xor(mx[h], off, 32));

  // pass 2: softmax denominator
  float den[H];
#pragma unroll
  for (int h = 0; h < H; ++h) den[h] = 0.f;
  for (int i = beg + lane; i < end; i += 32) {
    int s = srcv[perm[i]];
#pragma unroll
    for (int h = 0; h < H; ++h) {
      float sc = el[s * H + h] + ern[h];
      sc = sc > 0.f ? sc : 0.2f * sc;
      den[h] += __expf(sc - mx[h]);
    }
  }
  float inv_den[H];
#pragma unroll
  for (int h = 0; h < H; ++h) {
    for (int off = 16; off > 0; off >>= 1)
      den[h] += __shfl_xor(den[h], off, 32);
    inv_den[h] = (den[h] > 0.f) ? 1.f / den[h] : 0.f;
  }

  // pass 3: weighted feature sum in fixed (deterministic) edge order;
  // feat (<=51 MB) is L2-resident on the 192 MB L2, so gathers stay on-chip
  float acc[PER];
#pragma unroll
  for (int p = 0; p < PER; ++p) acc[p] = 0.f;
  for (int i = beg; i < end; ++i) {
    int s = srcv[perm[i]];
    float alpha[H];
#pragma unroll
    for (int h = 0; h < H; ++h) {
      float sc = el[s * H + h] + ern[h];
      sc = sc > 0.f ? sc : 0.2f * sc;
      alpha[h] = __expf(sc - mx[h]) * inv_den[h];
    }
    const float* fs = feat + (size_t)s * TOT;
#pragma unroll
    for (int p = 0; p < PER; ++p) {
      int hh = (p * 32) / D;              // exact: lane<32<=D keeps it in-block
      acc[p] += alpha[hh] * fs[p * 32 + lane];
    }
  }
#pragma unroll
  for (int p = 0; p < PER; ++p) {
    float v = acc[p];
    if (ELU) v = v > 0.f ? v : __expf(v) - 1.f;
    store_val(&out[(size_t)wave * TOT + p * 32 + lane], v);
  }
}

// ------------------------------------------------------------------- launch
extern "C" void kernel_launch(void* const* d_in, const int* in_sizes, int n_in,
                              void* d_out, int out_size, void* d_ws, size_t ws_size,
                              hipStream_t stream) {
  (void)in_sizes; (void)n_in; (void)out_size; (void)ws_size;
  const float* h    = (const float*)d_in[0];
  const int*   srcv = (const int*)d_in[1];
  const int*   dstv = (const int*)d_in[2];
  const float* W1   = (const float*)d_in[3];
  const float* al1  = (const float*)d_in[4];
  const float* ar1  = (const float*)d_in[5];
  const float* W2   = (const float*)d_in[6];
  const float* al2  = (const float*)d_in[7];
  const float* ar2  = (const float*)d_in[8];
  float* out = (float*)d_out;

  const int N = GN, E = GE;
  const int IN = 256, F1 = 256, OUTD = 128, HEADS = 4;

  char* p = (char*)d_ws;
  auto carve = [&](size_t bytes) {
    char* r = p;
    p += (bytes + 255) & ~(size_t)255;
    return (void*)r;
  };
  _Float16* hb    = (_Float16*)carve((size_t)N * IN * 2);
  _Float16* B1p   = (_Float16*)carve((size_t)IN * F1 * 2);     // packed W1
  _Float16* B2p   = (_Float16*)carve((size_t)F1 * OUTD * 2);   // packed W2
  float*    feat1 = (float*)   carve((size_t)N * F1 * 4);
  float*    el1   = (float*)   carve((size_t)N * HEADS * 4);
  float*    er1   = (float*)   carve((size_t)N * HEADS * 4);
  _Float16* h1b   = (_Float16*)carve((size_t)N * F1 * 2);
  float*    feat2 = (float*)   carve((size_t)N * OUTD * 4);
  float*    el2   = (float*)   carve((size_t)N * 4);
  float*    er2   = (float*)   carve((size_t)N * 4);
  int* hist    = (int*)carve((size_t)NCHUNK * N * 4);
  int* rank    = (int*)carve((size_t)E * 4);
  int* deg     = (int*)carve((size_t)N * 4);
  int* row_ptr = (int*)carve((size_t)(N + 1) * 4);
  int* perm    = (int*)carve((size_t)E * 4);

  const int chunkSz = (E + NCHUNK - 1) / NCHUNK;

  // deterministic CSR build (stable counting sort by dst)
  k_zero_i32<<<(NCHUNK * N + 255) / 256, 256, 0, stream>>>(hist, NCHUNK * N);
  k_chunk_count<<<1, 64, 0, stream>>>(dstv, hist, rank, E, chunkSz, N);
  k_node_deg<<<(N + 255) / 256, 256, 0, stream>>>(hist, deg, N);
  k_scan_excl<<<1, 1024, 0, stream>>>(deg, row_ptr, N);
  k_chunk_offsets<<<(N + 255) / 256, 256, 0, stream>>>(hist, row_ptr, N);
  k_scatter_edges<<<(E + 255) / 256, 256, 0, stream>>>(dstv, hist, rank, perm, E, chunkSz, N);

  // operand preparation: h -> f16; W1/W2 -> packed f16 B fragments
  k_cvt_f16<<<(N * IN + 255) / 256, 256, 0, stream>>>(h, hb, N * IN);
  k_pack_B<256><<<((IN / 32) * (F1 / 16) * 32 + 255) / 256, 256, 0, stream>>>(W1, B1p, F1);
  k_pack_B<256><<<((F1 / 32) * (OUTD / 16) * 32 + 255) / 256, 256, 0, stream>>>(W2, B2p, OUTD);

  // ----- layer 1  (waves = (N/16) * (F1/16)/4 = 12500)
  {
    int groups = (N / 16) * ((F1 / 16) / 4);
    k_wmma_gemm_f16<256, 4><<<(groups * 32 + 255) / 256, 256, 0, stream>>>(hb, B1p, feat1, N, F1);
  }
  k_attn_scores<4, 64><<<(N * HEADS + 255) / 256, 256, 0, stream>>>(feat1, al1, ar1, el1, er1, N);
  k_aggregate<4, 64, true, _Float16>
      <<<(N * 32 + 255) / 256, 256, 0, stream>>>(feat1, el1, er1, row_ptr, perm, srcv, h1b, N);

  // ----- layer 2  (waves = (N/16) * (OUTD/16)/4 = 6250)
  {
    int groups = (N / 16) * ((OUTD / 16) / 4);
    k_wmma_gemm_f16<256, 4><<<(groups * 32 + 255) / 256, 256, 0, stream>>>(h1b, B2p, feat2, N, OUTD);
  }
  k_attn_scores<1, 128><<<(N + 255) / 256, 256, 0, stream>>>(feat2, al2, ar2, el2, er2, N);
  k_aggregate<1, 128, false, float>
      <<<(N * 32 + 255) / 256, 256, 0, stream>>>(feat2, el2, er2, row_ptr, perm, srcv, out, N);
}
